// MyLSTM_86706799772013
// MI455X (gfx1250) — matrix operand order repaired and checked by reading, not compiled, not run on hardware
//
#include <hip/hip_runtime.h>

// ---------------------------------------------------------------------------
// 3-layer LSTM (PyTorch semantics) + linear + softmax(last step) for gfx1250.
// f16 WMMA (v_wmma_f32_16x16x32_f16), f32 accumulate.
//
// Pipeline per layer, per 128-step chunk:
//   [A] lstm_ingemm<D> : parallel WMMA GEMM  xg = inp @ W_ih^T + b (swizzled out)
//   [B] lstm_recur     : persistent 1-WG WMMA recurrence, W_hh B-frags in VGPRs,
//                        h double-buffered in LDS, c in accumulator registers,
//                        global traffic via base + 32-bit incremented offsets
//                        (keeps GLOBAL_ addrspace; no FLAT/DScnt coupling).
// Final: lstm_head     : [128]x[2] dot + softmax -> d_out (f32).
// ---------------------------------------------------------------------------

#define N_BATCH 128
#define L_SEQ   2048
#define H_DIM   128
#define G4      512          // 4*H
#define CHUNK   128          // timesteps per chunk
#define HP      136          // padded f16 row length in LDS (bank spread)

typedef __attribute__((ext_vector_type(16))) _Float16 v16h;
typedef __attribute__((ext_vector_type(8)))  _Float16 v8h;
typedef __attribute__((ext_vector_type(8)))  float    v8f;

__device__ __forceinline__ v16h cat16(v8h lo, v8h hi) {
  v16h r;
#pragma unroll
  for (int i = 0; i < 8; ++i) { r[i] = lo[i]; r[i + 8] = hi[i]; }
  return r;
}

__device__ __forceinline__ v8f wmma16(v16h a, v16h b, v8f c) {
  // (neg_a, A, neg_b, B, c_mod, C, reuse_a, reuse_b)
  return __builtin_amdgcn_wmma_f32_16x16x32_f16(false, a, false, b, (short)0, c,
                                                false, false);
}

__device__ __forceinline__ float sigf(float x) {
  return 1.f / (1.f + __expf(-x));
}
__device__ __forceinline__ float tanh_f(float x) {
  x = fminf(fmaxf(x, -15.f), 15.f);
  float e = __expf(-2.f * x);
  return (1.f - e) / (1.f + e);
}

// ---------------------------------------------------------------------------
// f32 -> f16 pack (grid-stride)
// ---------------------------------------------------------------------------
__global__ void f32_to_f16(const float* __restrict__ s, _Float16* __restrict__ d,
                           int n) {
  for (int i = blockIdx.x * blockDim.x + threadIdx.x; i < n;
       i += gridDim.x * blockDim.x)
    d[i] = (_Float16)s[i];
}

// ---------------------------------------------------------------------------
// Input GEMM: xg[tt, ctile, mtile, lane*8..] = inp[n, t0+tt, :] @ W_ih^T + b
//   inp : f16 [N][L][D]   (D = 64 or 128, compile-time)
//   wih : f16 [512][D] row-major
//   xg  : f16 swizzled: tile base = ((tt*32 + ctile)*8 + mtile)*256, C-lane order
// grid = (CHUNK, 4), block = 256 (8 waves); wave -> ctile, loops 8 m-tiles.
// ---------------------------------------------------------------------------
template <int D>
__global__ __launch_bounds__(256) void lstm_ingemm(
    const _Float16* __restrict__ inp,
    const _Float16* __restrict__ wih, const float* __restrict__ bias,
    _Float16* __restrict__ xg, int t0) {
  constexpr int NKS = D / 32;                 // K-steps of 32 (2 or 4)
  const int lane = threadIdx.x & 31;
  const int wv   = threadIdx.x >> 5;
  const int l15  = lane & 15;
  const int hi   = lane >> 4;                 // 0: lanes 0-15, 1: lanes 16-31
  const int tt   = blockIdx.x;
  const int t    = t0 + tt;
  const int ctile = blockIdx.y * 8 + wv;      // 0..31
  const int col   = ctile * 16 + l15;         // gate column 0..511

  // B fragments: lane holds K 8*hi..+7 and 16+8*hi..+7 for its column.
  v16h bf[NKS];
#pragma unroll
  for (int ks = 0; ks < NKS; ++ks) {
    const _Float16* bp = wih + (size_t)col * D + ks * 32 + hi * 8;
    bf[ks] = cat16(*(const v8h*)bp, *(const v8h*)(bp + 16));
  }
  const float bval = bias[col];

#pragma unroll
  for (int mt = 0; mt < 8; ++mt) {
    const int n = mt * 16 + l15;              // batch row for this lane
    const _Float16* ap = inp + ((size_t)n * L_SEQ + t) * D;
    v8f acc = {};
#pragma unroll
    for (int ks = 0; ks < NKS; ++ks) {
      const _Float16* a0 = ap + ks * 32 + hi * 8;   // K pairs: lo half
      v16h af = cat16(*(const v8h*)a0, *(const v8h*)(a0 + 16));
      acc = wmma16(af, bf[ks], acc);
    }
    v8h out;
#pragma unroll
    for (int v = 0; v < 8; ++v) out[v] = (_Float16)(acc[v] + bval);
    *(v8h*)(xg + ((size_t)((tt * 32 + ctile) * 8 + mt)) * 256 + lane * 8) = out;
  }
}

// ---------------------------------------------------------------------------
// Recurrence: one persistent workgroup (512 thr = 16 waves) per chunk.
//   gates = h_{t-1} @ W_hh^T  (WMMA, K=128) + xg_t ; LSTM cell update.
// Wave w: hc = w&7 (h-column tile), m-tiles {w>>3, +2, +4, +6}.
// W_hh B-fragments (4 gates x 4 ksteps) live in VGPRs for the whole chunk.
// h double-buffered in LDS; c lives in registers; one barrier per step.
// xg/z addressed as global-base + 32-bit offset (incremented per step) so the
// backend keeps GLOBAL_LOAD/STORE encodings with immediate per-u/g/v deltas.
// ---------------------------------------------------------------------------
__global__ __launch_bounds__(512, 1) void lstm_recur(
    const _Float16* __restrict__ whh_g,   // f16 [512][128]
    const _Float16* __restrict__ xg,      // f16 swizzled chunk
    _Float16* __restrict__ z,             // f16 [N][L][H] output hidden seq
    _Float16* __restrict__ hstate,        // f16 [128][128] carry
    float* __restrict__ cstate,           // f32 [128][128] carry
    int t0, int first) {
  extern __shared__ char smem[];
  _Float16* whh = (_Float16*)smem;                      // [512][HP]
  _Float16* hb  = (_Float16*)(smem + 512 * HP * 2);     // [2][128][HP]

  const int tid  = threadIdx.x;
  const int lane = tid & 31;
  const int wv   = tid >> 5;
  const int l15  = lane & 15;
  const int hi   = lane >> 4;
  const int hc   = wv & 7;        // h-column tile 0..7
  const int mt0  = wv >> 3;       // 0 or 1

  // Stage W_hh into LDS (padded rows), vectorized b128 copies.
  for (int s = tid; s < 512 * 16; s += 512) {
    const int r = s >> 4, k8 = (s & 15) * 8;
    *(v8h*)(whh + r * HP + k8) = *(const v8h*)(whh_g + r * 128 + k8);
  }
  // Init h buffer 0 with carried state (or zeros on first chunk).
  for (int s = tid; s < 128 * 16; s += 512) {
    const int r = s >> 4, k8 = (s & 15) * 8;
    v8h v;
    if (first) {
#pragma unroll
      for (int i = 0; i < 8; ++i) v[i] = (_Float16)0.f;
    } else {
      v = *(const v8h*)(hstate + r * 128 + k8);
    }
    *(v8h*)(hb + r * HP + k8) = v;
  }
  __syncthreads();

  // Pre-build W_hh B-fragments for this wave's hc: constant across all steps.
  v16h bf[4][4];
#pragma unroll
  for (int g = 0; g < 4; ++g) {
    const int col = g * 128 + hc * 16 + l15;  // gate column in [0,512)
#pragma unroll
    for (int ks = 0; ks < 4; ++ks) {
      const _Float16* bp = whh + col * HP + ks * 32 + hi * 8;
      bf[g][ks] = cat16(*(const v8h*)bp, *(const v8h*)(bp + 16));
    }
  }

  // Cell state in registers (C-layout: row = mt*16 + v + 8*hi, col fixed).
  const int ccol = hc * 16 + l15;
  float c[4][8];
#pragma unroll
  for (int u = 0; u < 4; ++u) {
    const int mt = mt0 + 2 * u;
#pragma unroll
    for (int v = 0; v < 8; ++v) {
      const int row = mt * 16 + v + 8 * hi;
      c[u][v] = first ? 0.f : cstate[row * 128 + ccol];
    }
  }

  // 32-bit offsets (halves) off the global bases.
  //   xg: per-step stride 65536; per-u delta u*512 (imm); per-gate 16384 (imm)
  //   z : per-step stride H_DIM; per-v delta v*L*H (imm); one offset per u
  int xgoff = (hc * 8 + mt0) * 256 + lane * 8;
  int zoff[4];
#pragma unroll
  for (int u = 0; u < 4; ++u) {
    const int mt = mt0 + 2 * u;
    zoff[u] = ((mt * 16 + 8 * hi) * L_SEQ + t0) * H_DIM + ccol;
  }
  int hoff = 0;                       // LDS double-buffer toggle (halves)
  const int HBUF = 128 * HP;

  for (int tt = 0; tt < CHUNK; ++tt) {
    __syncthreads();  // h_{t-1} visible in hb[hoff]
    const _Float16* hcur = hb + hoff;
    _Float16* hnxt = hb + (hoff ^ HBUF);
    hoff ^= HBUF;

#pragma unroll
    for (int u = 0; u < 4; ++u) {
      const int mt = mt0 + 2 * u;
      v8f acc0 = {}, acc1 = {}, acc2 = {}, acc3 = {};
      const int arow = mt * 16 + l15;
#pragma unroll
      for (int ks = 0; ks < 4; ++ks) {
        const _Float16* ap = hcur + arow * HP + ks * 32 + hi * 8;
        v16h af = cat16(*(const v8h*)ap, *(const v8h*)(ap + 16));
        acc0 = wmma16(af, bf[0][ks], acc0);
        acc1 = wmma16(af, bf[1][ks], acc1);
        acc2 = wmma16(af, bf[2][ks], acc2);
        acc3 = wmma16(af, bf[3][ks], acc3);
      }
      // Precomputed xg (bias folded in) — same swizzled C-lane layout.
      // u/g deltas are compile-time -> folded into GLOBAL_LOAD immediates.
      const v8h x0 = *(const v8h*)(xg + xgoff + u * 512);
      const v8h x1 = *(const v8h*)(xg + xgoff + u * 512 + 16384);
      const v8h x2 = *(const v8h*)(xg + xgoff + u * 512 + 32768);
      const v8h x3 = *(const v8h*)(xg + xgoff + u * 512 + 49152);

      _Float16* hrow = hnxt + (mt * 16 + 8 * hi) * HP + ccol;
      const int zo = zoff[u];
      zoff[u] += H_DIM;
      // LSTM cell: pure VALU on the accumulator layout.
#pragma unroll
      for (int v = 0; v < 8; ++v) {
        const float ig = sigf(acc0[v] + (float)x0[v]);
        const float fg = sigf(acc1[v] + (float)x1[v]);
        const float gg = tanh_f(acc2[v] + (float)x2[v]);
        const float og = sigf(acc3[v] + (float)x3[v]);
        const float cn = fg * c[u][v] + ig * gg;
        c[u][v] = cn;
        const _Float16 hh = (_Float16)(og * tanh_f(cn));
        hrow[v * HP] = hh;                    // LDS, immediate offsets
        z[zo + v * (L_SEQ * H_DIM)] = hh;     // global, immediate offsets
      }
    }
    xgoff += 65536;   // next timestep's xg slab
  }
  __syncthreads();

  // Save carries (CHUNK is even -> final h is in buffer 0).
  for (int s = tid; s < 128 * 16; s += 512) {
    const int r = s >> 4, k8 = (s & 15) * 8;
    *(v8h*)(hstate + r * 128 + k8) = *(const v8h*)(hb + r * HP + k8);
  }
#pragma unroll
  for (int u = 0; u < 4; ++u) {
    const int mt = mt0 + 2 * u;
#pragma unroll
    for (int v = 0; v < 8; ++v)
      cstate[(mt * 16 + v + 8 * hi) * 128 + ccol] = c[u][v];
  }
}

// ---------------------------------------------------------------------------
// Head: out[n,:] = softmax(z3[n, L-1, :] @ W_lin^T + b_lin)
// ---------------------------------------------------------------------------
__global__ void lstm_head(const _Float16* __restrict__ z,
                          const float* __restrict__ wlin,
                          const float* __restrict__ blin,
                          float* __restrict__ out) {
  const int n = blockIdx.x * blockDim.x + threadIdx.x;
  if (n >= N_BATCH) return;
  const _Float16* zp = z + ((size_t)n * L_SEQ + (L_SEQ - 1)) * H_DIM;
  float a0 = blin[0], a1 = blin[1];
  for (int k = 0; k < H_DIM; ++k) {
    const float zv = (float)zp[k];
    a0 += zv * wlin[k];
    a1 += zv * wlin[H_DIM + k];
  }
  const float m = fmaxf(a0, a1);
  const float e0 = __expf(a0 - m), e1 = __expf(a1 - m);
  const float s = e0 + e1;
  out[n * 2 + 0] = e0 / s;
  out[n * 2 + 1] = e1 / s;
}

// ---------------------------------------------------------------------------
extern "C" void kernel_launch(void* const* d_in, const int* in_sizes, int n_in,
                              void* d_out, int out_size, void* d_ws,
                              size_t ws_size, hipStream_t stream) {
  (void)in_sizes; (void)n_in; (void)out_size; (void)ws_size;

  const float* x     = (const float*)d_in[0];
  const float* w_ih[3] = {(const float*)d_in[1], (const float*)d_in[4],
                          (const float*)d_in[7]};
  const float* w_hh[3] = {(const float*)d_in[2], (const float*)d_in[5],
                          (const float*)d_in[8]};
  const float* bias[3] = {(const float*)d_in[3], (const float*)d_in[6],
                          (const float*)d_in[9]};
  const float* w_lin = (const float*)d_in[10];
  const float* b_lin = (const float*)d_in[11];

  // Workspace carve-out (256B aligned slabs).
  char* ws = (char*)d_ws;
  size_t off = 0;
  auto carve = [&](size_t bytes) {
    char* p = ws + off;
    off += (bytes + 255) & ~(size_t)255;
    return p;
  };
  _Float16* xh  = (_Float16*)carve((size_t)N_BATCH * L_SEQ * 64 * 2);   // 32 MB
  _Float16* zA  = (_Float16*)carve((size_t)N_BATCH * L_SEQ * H_DIM * 2);// 64 MB
  _Float16* zB  = (_Float16*)carve((size_t)N_BATCH * L_SEQ * H_DIM * 2);// 64 MB
  _Float16* xg  = (_Float16*)carve((size_t)CHUNK * 32 * 8 * 256 * 2);   // 16 MB
  _Float16* wihp[3], *whhp[3];
  wihp[0] = (_Float16*)carve((size_t)G4 * 64 * 2);
  whhp[0] = (_Float16*)carve((size_t)G4 * H_DIM * 2);
  wihp[1] = (_Float16*)carve((size_t)G4 * H_DIM * 2);
  whhp[1] = (_Float16*)carve((size_t)G4 * H_DIM * 2);
  wihp[2] = (_Float16*)carve((size_t)G4 * H_DIM * 2);
  whhp[2] = (_Float16*)carve((size_t)G4 * H_DIM * 2);
  _Float16* hstate = (_Float16*)carve((size_t)128 * 128 * 2);
  float*    cstate = (float*)carve((size_t)128 * 128 * 4);

  // ---- pack f32 -> f16 ----
  f32_to_f16<<<2048, 256, 0, stream>>>(x, xh, N_BATCH * L_SEQ * 64);
  f32_to_f16<<<32, 256, 0, stream>>>(w_ih[0], wihp[0], G4 * 64);
  f32_to_f16<<<64, 256, 0, stream>>>(w_hh[0], whhp[0], G4 * H_DIM);
  f32_to_f16<<<64, 256, 0, stream>>>(w_ih[1], wihp[1], G4 * H_DIM);
  f32_to_f16<<<64, 256, 0, stream>>>(w_hh[1], whhp[1], G4 * H_DIM);
  f32_to_f16<<<64, 256, 0, stream>>>(w_ih[2], wihp[2], G4 * H_DIM);
  f32_to_f16<<<64, 256, 0, stream>>>(w_hh[2], whhp[2], G4 * H_DIM);

  const size_t smem = 512 * HP * 2 + 2 * 128 * HP * 2;  // 208,896 B (< 320 KB)
  const _Float16* inp[3] = {xh, zA, zB};
  _Float16* outz[3]      = {zA, zB, zA};

  for (int layer = 0; layer < 3; ++layer) {
    for (int ck = 0; ck < L_SEQ / CHUNK; ++ck) {
      const int t0 = ck * CHUNK;
      if (layer == 0) {
        lstm_ingemm<64><<<dim3(CHUNK, 4), 256, 0, stream>>>(
            inp[layer], wihp[layer], bias[layer], xg, t0);
      } else {
        lstm_ingemm<128><<<dim3(CHUNK, 4), 256, 0, stream>>>(
            inp[layer], wihp[layer], bias[layer], xg, t0);
      }
      lstm_recur<<<1, 512, smem, stream>>>(whhp[layer], xg, outz[layer],
                                           hstate, cstate, t0, ck == 0);
    }
  }

  lstm_head<<<1, 128, 0, stream>>>(zA, w_lin, b_lin, (float*)d_out);
}
